// AutoEpsilonSingleSelfAttention_73701638799897
// MI455X (gfx1250) — matrix operand order repaired and checked by reading, not compiled, not run on hardware
//
#include <hip/hip_runtime.h>
#include <hip/hip_bf16.h>
#include <math.h>

// ---------------------------------------------------------------------------
// AutoEpsilonSingleSelfAttention for MI455X (gfx1250, wave32, WMMA)
//   h0 = LN(gelu(x @ W0 + b0))            -> f16, materialized (64MB)
//   s  = LN(gelu(h0 @ W1 + b1)) @ Wa + ba -> fused, h1 never materialized
//   alpha = softmax_S(s); eps = sigmoid(alpha @ We + be)
//   greaters = alpha > eps; candidate = greaters ? x : 0
// 32 rows per block (2 M-tiles/wave) to halve L2 weight traffic; branch-free
// GELU via exp (0.5*(1+tanh(u)) == sigmoid(2u), exact identity).
// ---------------------------------------------------------------------------

typedef __attribute__((ext_vector_type(16))) _Float16 v16h;
typedef __attribute__((ext_vector_type(8)))  float    v8f;

#define BB    64
#define SS    512
#define DD    1024
#define NROWS (BB*SS)      // 32768
#define KDIM  1024
#define NDIM  1024
#define KSTEP 32
#define MT    2            // M tiles per block
#define RPB   (MT*16)      // 32 rows per block
#define LDA   40           // padded LDS row stride for A tile (f16 elements)

// gelu_tanh(x) = 0.5x(1+tanh(u)) = x * sigmoid(2u), u = c*(x + 0.044715 x^3)
__device__ __forceinline__ float gelu_fast(float x) {
    float u = x + 0.044715f * x * x * x;
    float z = __expf(-1.5957691216057308f * u);   // exp(-2*0.79788456*u)
    return x * __builtin_amdgcn_rcpf(1.0f + z);
}

__device__ __forceinline__ float sigmoid_fast(float x) {
    return __builtin_amdgcn_rcpf(1.0f + __expf(-x));
}

// A fragment (16x32 f16): lane half0 -> K{0..7,16..23}, half1 -> K{8..15,24..31}
__device__ __forceinline__ v16h frag_a_ld(const _Float16* rowp, int half) {
    v16h r;
    const _Float16* p0 = rowp + half * 8;        // elems 0..7
    const _Float16* p1 = rowp + 16 + half * 8;   // elems 8..15
#pragma unroll
    for (int i = 0; i < 8; ++i) { r[i] = p0[i]; r[i + 8] = p1[i]; }
    return r;
}

// B fragment (32x16 f16, stored as Wt[N][K]): half0 -> K{0..7,16..23}, half1 -> K{8..15,24..31}
__device__ __forceinline__ v16h frag_b_ld(const _Float16* rowp, int half) {
    v16h r;
    const _Float16* p0 = rowp + half * 8;
    const _Float16* p1 = rowp + 16 + half * 8;
#pragma unroll
    for (int i = 0; i < 8; ++i) { r[i] = p0[i]; r[i + 8] = p1[i]; }
    return r;
}

// --------------------------- weight transpose/convert ----------------------
__global__ __launch_bounds__(256)
void transpose_f32_to_f16(const float* __restrict__ W, _Float16* __restrict__ Wt) {
    __shared__ float tile[32][33];
    int n0 = blockIdx.x * 32, k0 = blockIdx.y * 32;
    int tx = threadIdx.x & 31, ty = threadIdx.x >> 5;   // ty in 0..7
#pragma unroll
    for (int j = 0; j < 32; j += 8)
        tile[ty + j][tx] = W[(size_t)(k0 + ty + j) * NDIM + n0 + tx];
    __syncthreads();
#pragma unroll
    for (int j = 0; j < 32; j += 8)
        Wt[(size_t)(n0 + ty + j) * KDIM + k0 + tx] = (_Float16)tile[tx][ty + j];
}

// ------------------- kernel 1: X(f32) @ W0 -> gelu -> LN -> H0(f16) --------
__global__ __launch_bounds__(256)
void dense_gelu_ln_k1(const float* __restrict__ X, const _Float16* __restrict__ Wt,
                      const float* __restrict__ bias, _Float16* __restrict__ Hout) {
    __shared__ _Float16 lds_a[RPB * LDA];
    __shared__ float red_s[RPB][8];
    __shared__ float red_q[RPB][8];
    __shared__ float st_mean[RPB], st_rstd[RPB];

    const int t    = threadIdx.x;
    const int wave = t >> 5;
    const int lane = t & 31;
    const int half = lane >> 4;
    const int lm   = lane & 15;
    const int row0 = blockIdx.x * RPB;

    v8f acc[MT][8];
#pragma unroll
    for (int g = 0; g < MT; ++g)
#pragma unroll
        for (int i = 0; i < 8; ++i) { v8f z = {}; acc[g][i] = z; }

    for (int kt = 0; kt < KDIM / KSTEP; ++kt) {
        __syncthreads();
        { // stage A tile: RPB rows x 32 K, f32 -> f16 (4 elems/thread)
            int e = t * 4, m = e >> 5, k = e & 31;
            const float* src = X + (size_t)(row0 + m) * KDIM + kt * KSTEP + k;
#pragma unroll
            for (int j = 0; j < 4; ++j)
                lds_a[m * LDA + k + j] = (_Float16)src[j];
        }
        __syncthreads();
        v16h a[MT];
#pragma unroll
        for (int g = 0; g < MT; ++g)
            a[g] = frag_a_ld(lds_a + (g * 16 + lm) * LDA, half);
#pragma unroll
        for (int tt = 0; tt < 8; ++tt) {
            int n = wave * 128 + tt * 16 + lm;
            const _Float16* bp = Wt + (size_t)n * KDIM + kt * KSTEP;
            v16h b = frag_b_ld(bp, half);
            if (kt + 1 < KDIM / KSTEP)
                __builtin_prefetch(bp + KSTEP, 0, 3);   // global_prefetch_b8
#pragma unroll
            for (int g = 0; g < MT; ++g)
                acc[g][tt] = __builtin_amdgcn_wmma_f32_16x16x32_f16(
                    false, a[g], false, b, (short)0, acc[g][tt], false, false);
        }
    }

    // epilogue: bias + gelu in-register, LN partials (row = g*16 + r + 8*half)
    float s[MT][8], q[MT][8];
#pragma unroll
    for (int g = 0; g < MT; ++g)
#pragma unroll
        for (int r = 0; r < 8; ++r) { s[g][r] = 0.f; q[g][r] = 0.f; }
#pragma unroll
    for (int tt = 0; tt < 8; ++tt) {
        int n = wave * 128 + tt * 16 + lm;
        float bv = bias[n];
#pragma unroll
        for (int g = 0; g < MT; ++g)
#pragma unroll
            for (int r = 0; r < 8; ++r) {
                float v = gelu_fast(acc[g][tt][r] + bv);
                acc[g][tt][r] = v;
                s[g][r] += v; q[g][r] += v * v;
            }
    }
#pragma unroll
    for (int g = 0; g < MT; ++g)
#pragma unroll
        for (int r = 0; r < 8; ++r)
            for (int off = 1; off < 16; off <<= 1) {   // reduce within each half
                s[g][r] += __shfl_xor(s[g][r], off, 32);
                q[g][r] += __shfl_xor(q[g][r], off, 32);
            }
    if (lm == 0) {
#pragma unroll
        for (int g = 0; g < MT; ++g)
#pragma unroll
            for (int r = 0; r < 8; ++r) {
                red_s[g * 16 + r + 8 * half][wave] = s[g][r];
                red_q[g * 16 + r + 8 * half][wave] = q[g][r];
            }
    }
    __syncthreads();
    if (t < RPB) {
        float ss = 0.f, qq = 0.f;
#pragma unroll
        for (int w = 0; w < 8; ++w) { ss += red_s[t][w]; qq += red_q[t][w]; }
        float mean = ss * (1.0f / 1024.0f);
        float var  = qq * (1.0f / 1024.0f) - mean * mean;
        st_mean[t] = mean;
        st_rstd[t] = rsqrtf(var + 1e-6f);
    }
    __syncthreads();
#pragma unroll
    for (int tt = 0; tt < 8; ++tt) {
        int n = wave * 128 + tt * 16 + lm;
#pragma unroll
        for (int g = 0; g < MT; ++g)
#pragma unroll
            for (int r = 0; r < 8; ++r) {
                int m = g * 16 + r + 8 * half;
                float v = (acc[g][tt][r] - st_mean[m]) * st_rstd[m];
                Hout[(size_t)(row0 + m) * NDIM + n] = (_Float16)v;
            }
    }
}

// ---- kernel 2: H0(f16) @ W1 -> gelu -> LN -> dot(Wa) -> scores (no H1) ----
__global__ __launch_bounds__(256)
void dense_gelu_ln_dot_k2(const _Float16* __restrict__ Hin, const _Float16* __restrict__ Wt,
                          const float* __restrict__ bias, const float* __restrict__ Wa,
                          const float* __restrict__ ba, float* __restrict__ scores) {
    __shared__ _Float16 lds_a[RPB * LDA];
    __shared__ float red_s[RPB][8];
    __shared__ float red_q[RPB][8];
    __shared__ float st_mean[RPB], st_rstd[RPB];

    const int t    = threadIdx.x;
    const int wave = t >> 5;
    const int lane = t & 31;
    const int half = lane >> 4;
    const int lm   = lane & 15;
    const int row0 = blockIdx.x * RPB;

    v8f acc[MT][8];
#pragma unroll
    for (int g = 0; g < MT; ++g)
#pragma unroll
        for (int i = 0; i < 8; ++i) { v8f z = {}; acc[g][i] = z; }

    for (int kt = 0; kt < KDIM / KSTEP; ++kt) {
        __syncthreads();
        { // stage A tile from f16 source: 4 f16 = 8B copy per thread
            int e = t * 4, m = e >> 5, k = e & 31;
            const _Float16* src = Hin + (size_t)(row0 + m) * KDIM + kt * KSTEP + k;
            *(uint2*)&lds_a[m * LDA + k] = *(const uint2*)src;
        }
        __syncthreads();
        v16h a[MT];
#pragma unroll
        for (int g = 0; g < MT; ++g)
            a[g] = frag_a_ld(lds_a + (g * 16 + lm) * LDA, half);
#pragma unroll
        for (int tt = 0; tt < 8; ++tt) {
            int n = wave * 128 + tt * 16 + lm;
            const _Float16* bp = Wt + (size_t)n * KDIM + kt * KSTEP;
            v16h b = frag_b_ld(bp, half);
            if (kt + 1 < KDIM / KSTEP)
                __builtin_prefetch(bp + KSTEP, 0, 3);   // global_prefetch_b8
#pragma unroll
            for (int g = 0; g < MT; ++g)
                acc[g][tt] = __builtin_amdgcn_wmma_f32_16x16x32_f16(
                    false, a[g], false, b, (short)0, acc[g][tt], false, false);
        }
    }

    float s[MT][8], q[MT][8];
#pragma unroll
    for (int g = 0; g < MT; ++g)
#pragma unroll
        for (int r = 0; r < 8; ++r) { s[g][r] = 0.f; q[g][r] = 0.f; }
    float wv[8];
#pragma unroll
    for (int tt = 0; tt < 8; ++tt) {
        int n = wave * 128 + tt * 16 + lm;
        wv[tt] = Wa[n];
        float bv = bias[n];
#pragma unroll
        for (int g = 0; g < MT; ++g)
#pragma unroll
            for (int r = 0; r < 8; ++r) {
                float v = gelu_fast(acc[g][tt][r] + bv);
                acc[g][tt][r] = v;
                s[g][r] += v; q[g][r] += v * v;
            }
    }
#pragma unroll
    for (int g = 0; g < MT; ++g)
#pragma unroll
        for (int r = 0; r < 8; ++r)
            for (int off = 1; off < 16; off <<= 1) {
                s[g][r] += __shfl_xor(s[g][r], off, 32);
                q[g][r] += __shfl_xor(q[g][r], off, 32);
            }
    if (lm == 0) {
#pragma unroll
        for (int g = 0; g < MT; ++g)
#pragma unroll
            for (int r = 0; r < 8; ++r) {
                red_s[g * 16 + r + 8 * half][wave] = s[g][r];
                red_q[g * 16 + r + 8 * half][wave] = q[g][r];
            }
    }
    __syncthreads();
    if (t < RPB) {
        float ss = 0.f, qq = 0.f;
#pragma unroll
        for (int w = 0; w < 8; ++w) { ss += red_s[t][w]; qq += red_q[t][w]; }
        float mean = ss * (1.0f / 1024.0f);
        float var  = qq * (1.0f / 1024.0f) - mean * mean;
        st_mean[t] = mean;
        st_rstd[t] = rsqrtf(var + 1e-6f);
    }
    __syncthreads();
    // per-row dot with Wa over normalized values, straight from registers
    float d[MT][8];
#pragma unroll
    for (int g = 0; g < MT; ++g)
#pragma unroll
        for (int r = 0; r < 8; ++r) d[g][r] = 0.f;
#pragma unroll
    for (int tt = 0; tt < 8; ++tt) {
#pragma unroll
        for (int g = 0; g < MT; ++g)
#pragma unroll
            for (int r = 0; r < 8; ++r) {
                int m = g * 16 + r + 8 * half;
                d[g][r] += (acc[g][tt][r] - st_mean[m]) * st_rstd[m] * wv[tt];
            }
    }
#pragma unroll
    for (int g = 0; g < MT; ++g)
#pragma unroll
        for (int r = 0; r < 8; ++r)
            for (int off = 1; off < 16; off <<= 1)
                d[g][r] += __shfl_xor(d[g][r], off, 32);
    if (lm == 0) {
#pragma unroll
        for (int g = 0; g < MT; ++g)
#pragma unroll
            for (int r = 0; r < 8; ++r)
                red_s[g * 16 + r + 8 * half][wave] = d[g][r];
    }
    __syncthreads();
    if (t < RPB) {
        float sc = 0.f;
#pragma unroll
        for (int w = 0; w < 8; ++w) sc += red_s[t][w];
        scores[row0 + t] = sc + ba[0];
    }
}

// --------------------------- softmax over S per batch ----------------------
__global__ __launch_bounds__(256)
void softmax_rows(const float* __restrict__ scores, float* __restrict__ alpha) {
    __shared__ float red[8];
    int b = blockIdx.x, t = threadIdx.x;
    const float* srow = scores + (size_t)b * SS;
    float v0 = srow[t], v1 = srow[t + 256];
    float m = fmaxf(v0, v1);
    for (int off = 1; off < 32; off <<= 1) m = fmaxf(m, __shfl_xor(m, off, 32));
    if ((t & 31) == 0) red[t >> 5] = m;
    __syncthreads();
    float gm = red[0];
#pragma unroll
    for (int i = 1; i < 8; ++i) gm = fmaxf(gm, red[i]);
    float e0 = __expf(v0 - gm), e1 = __expf(v1 - gm);
    float ssum = e0 + e1;
    for (int off = 1; off < 32; off <<= 1) ssum += __shfl_xor(ssum, off, 32);
    __syncthreads();
    if ((t & 31) == 0) red[t >> 5] = ssum;
    __syncthreads();
    float gs = 0.f;
#pragma unroll
    for (int i = 0; i < 8; ++i) gs += red[i];
    float inv = 1.0f / gs;
    alpha[(size_t)b * SS + t]       = e0 * inv;
    alpha[(size_t)b * SS + t + 256] = e1 * inv;
}

// ---------------- epsilon = sigmoid(alpha @ We + be); greaters -------------
__global__ __launch_bounds__(256)
void eps_greater(const float* __restrict__ alpha, const float* __restrict__ We,
                 const float* __restrict__ be, float* __restrict__ gre) {
    __shared__ float arow[SS];
    int b = blockIdx.x, t = threadIdx.x;
    arow[t]       = alpha[(size_t)b * SS + t];
    arow[t + 256] = alpha[(size_t)b * SS + t + 256];
    __syncthreads();
#pragma unroll
    for (int c0 = 0; c0 < 2; ++c0) {
        int col = t + c0 * 256;
        float acc = 0.f;
        for (int k = 0; k < SS; ++k) acc += arow[k] * We[(size_t)k * SS + col];
        float eps = sigmoid_fast(acc + be[col]);
        gre[(size_t)b * SS + col] = (arow[col] > eps) ? 1.0f : 0.0f;
    }
}

// ----------------------- candidate = greaters ? x : 0 ----------------------
__global__ __launch_bounds__(256)
void mask_copy(const float* __restrict__ X, const float* __restrict__ gre,
               float* __restrict__ out) {
    int bs = blockIdx.x, t = threadIdx.x;
    float f = gre[bs];
    const float4* xs = (const float4*)(X + (size_t)bs * DD);
    float4* os = (float4*)(out + (size_t)bs * DD);
    float4 v = xs[t];
    if (f < 0.5f) { v.x = 0.f; v.y = 0.f; v.z = 0.f; v.w = 0.f; }
    os[t] = v;
}

// ---------------------------------------------------------------------------
extern "C" void kernel_launch(void* const* d_in, const int* in_sizes, int n_in,
                              void* d_out, int out_size, void* d_ws, size_t ws_size,
                              hipStream_t stream) {
    (void)in_sizes; (void)n_in; (void)out_size; (void)ws_size;
    const float* x  = (const float*)d_in[0];
    const float* W0 = (const float*)d_in[1];
    const float* b0 = (const float*)d_in[2];
    const float* W1 = (const float*)d_in[3];
    const float* b1 = (const float*)d_in[4];
    const float* Wa = (const float*)d_in[5];
    const float* ba = (const float*)d_in[6];
    const float* We = (const float*)d_in[7];
    const float* be = (const float*)d_in[8];

    float* out  = (float*)d_out;
    float* gre  = out;              // [B*S] greaters as 0/1 float
    float* cand = out + NROWS;      // [B*S*D]

    char* ws = (char*)d_ws;
    _Float16* W0t = (_Float16*)ws;  ws += (size_t)KDIM * NDIM * sizeof(_Float16);
    _Float16* W1t = (_Float16*)ws;  ws += (size_t)KDIM * NDIM * sizeof(_Float16);
    _Float16* H0  = (_Float16*)ws;  ws += (size_t)NROWS * NDIM * sizeof(_Float16);
    float* scores = (float*)ws;     ws += (size_t)NROWS * sizeof(float);
    float* alpha  = (float*)ws;     ws += (size_t)NROWS * sizeof(float);

    dim3 tgrid(NDIM / 32, KDIM / 32);
    transpose_f32_to_f16<<<tgrid, 256, 0, stream>>>(W0, W0t);
    transpose_f32_to_f16<<<tgrid, 256, 0, stream>>>(W1, W1t);

    dense_gelu_ln_k1<<<NROWS / RPB, 256, 0, stream>>>(x, W0t, b0, H0);
    dense_gelu_ln_dot_k2<<<NROWS / RPB, 256, 0, stream>>>(H0, W1t, b1, Wa, ba, scores);
    softmax_rows<<<BB, 256, 0, stream>>>(scores, alpha);
    eps_greater<<<BB, 256, 0, stream>>>(alpha, We, be, gre);
    mask_copy<<<NROWS, 256, 0, stream>>>(x, gre, cand);
}
